// ScaledDotProductAttention_8074538516733
// MI455X (gfx1250) — matrix operand order repaired and checked by reading, not compiled, not run on hardware
//
#include <hip/hip_runtime.h>

// Causal scaled-dot-product attention, fp32 in/out, MI455X / gfx1250.
// B=8, S=2048, D=256. d_out = out (B,S,D) ++ attention (B,S,S).
//
// QK^T uses bf16x3 error-compensated WMMA (V_WMMA_F32_16X16X32_BF16, fast pipe):
//   S = Qhi*Khi + Qhi*Klo + Qlo*Khi   (f32 accumulate, ~2^-18 rel err on logits)
// P*V uses exact-precision V_WMMA_F32_16X16X4_F32.

typedef __attribute__((ext_vector_type(2)))  float  v2f;
typedef __attribute__((ext_vector_type(8)))  float  v8f;
typedef __attribute__((ext_vector_type(8)))  __bf16 v8bf;
typedef __attribute__((ext_vector_type(16))) __bf16 v16bf;

#define BATCH 8
#define SEQ   2048
#define DIM   256
#define NW    8            // waves per block (wave32)
#define SS_STRIDE 2052     // 2048 + 4 pad -> conflict-free LDS strip
#define SCALE 0.0625f      // 1/sqrt(256)

#define SHUF16(a, b) __builtin_shufflevector((a), (b), 0, 1, 2, 3, 4, 5, 6, 7, \
                                             8, 9, 10, 11, 12, 13, 14, 15)

__device__ __forceinline__ unsigned bf16_rne_bits(unsigned u) {
    // round-to-nearest-even f32 -> bf16 (inputs are finite normals)
    return (u + 0x7FFFu + ((u >> 16) & 1u)) >> 16;
}

// ---------------------------------------------------------------------------
// Pre-pass: split f32 tensor into bf16 hi/lo planes (row-major, packed pairs).
// ---------------------------------------------------------------------------
__global__ __launch_bounds__(256)
void split_hi_lo_bf16(const float* __restrict__ src,
                      unsigned* __restrict__ hi2,
                      unsigned* __restrict__ lo2,
                      int n2)    // count of float2 elements
{
    int i = blockIdx.x * blockDim.x + threadIdx.x;
    if (i >= n2) return;
    const float2 x = ((const float2*)src)[i];
    const unsigned h0 = bf16_rne_bits(__float_as_uint(x.x));
    const unsigned h1 = bf16_rne_bits(__float_as_uint(x.y));
    const float r0 = x.x - __uint_as_float(h0 << 16);
    const float r1 = x.y - __uint_as_float(h1 << 16);
    const unsigned l0 = bf16_rne_bits(__float_as_uint(r0));
    const unsigned l1 = bf16_rne_bits(__float_as_uint(r1));
    hi2[i] = h0 | (h1 << 16);
    lo2[i] = l0 | (l1 << 16);
}

// ---------------------------------------------------------------------------
// Main attention kernel: 1 block = 16 query rows, 8 wave32s.
// ---------------------------------------------------------------------------
__global__ __launch_bounds__(256)
void sdpa_causal_wmma(const unsigned short* __restrict__ Qhi,
                      const unsigned short* __restrict__ Qlo,
                      const unsigned short* __restrict__ Khi,
                      const unsigned short* __restrict__ Klo,
                      const float* __restrict__ V,
                      float* __restrict__ out,
                      float* __restrict__ attn)
{
    extern __shared__ float smem[];
    float* sS     = smem;                     // [16][SS_STRIDE] score/prob strip
    float* sRed   = smem + 16 * SS_STRIDE;    // [16][16] reduction scratch
    float* rowMax = sRed + 256;               // [16]
    float* rowInv = rowMax + 16;              // [16]

    const int tid  = threadIdx.x;
    const int lane = tid & 31;
    const int wave = tid >> 5;
    const int m    = lane & 15;   // A-row / B-col / C-col index
    const int kh   = lane >> 4;   // lane-half select

    const int qTile = blockIdx.x & (SEQ / 16 - 1);
    const int b     = blockIdx.x / (SEQ / 16);
    const int q0    = qTile * 16;

    // ------------- Phase 1: S = (Q K^T) * scale via bf16x3 WMMA -------------
    {
        const size_t qrow = ((size_t)b * SEQ + q0 + m) * DIM;
        const unsigned short* qhp = Qhi + qrow;
        const unsigned short* qlp = Qlo + qrow;

        for (int kt = wave; kt <= qTile; kt += NW) {
            const int k0 = kt * 16;
            const size_t krow = ((size_t)b * SEQ + k0 + m) * DIM;
            const unsigned short* khp = Khi + krow;
            const unsigned short* klp = Klo + krow;

            v8f acc = {0.f, 0.f, 0.f, 0.f, 0.f, 0.f, 0.f, 0.f};
            #pragma unroll
            for (int c = 0; c < DIM / 32; ++c) {
                // A (16x32 bf16) layout: VGPR0-3 <- K = 8*kh..+7, VGPR4-7 <- K = 16+8*kh..+7
                const int oA0 = c * 32 + 8 * kh;
                const int oA1 = oA0 + 16;
                // B (32x16 bf16) layout: same as A, K = 8*kh..+7 then 16+8*kh..+7
                const int oB  = c * 32 + 8 * kh;

                v8bf qh0 = *(const v8bf*)(qhp + oA0);
                v8bf qh1 = *(const v8bf*)(qhp + oA1);
                v8bf ql0 = *(const v8bf*)(qlp + oA0);
                v8bf ql1 = *(const v8bf*)(qlp + oA1);
                v16bf ah = SHUF16(qh0, qh1);
                v16bf al = SHUF16(ql0, ql1);
                v16bf bh = SHUF16(*(const v8bf*)(khp + oB), *(const v8bf*)(khp + oB + 16));
                v16bf bl = SHUF16(*(const v8bf*)(klp + oB), *(const v8bf*)(klp + oB + 16));

                acc = __builtin_amdgcn_wmma_f32_16x16x32_bf16(
                    false, ah, false, bh, (short)0, acc, false, false);
                acc = __builtin_amdgcn_wmma_f32_16x16x32_bf16(
                    false, ah, false, bl, (short)0, acc, false, false);
                acc = __builtin_amdgcn_wmma_f32_16x16x32_bf16(
                    false, al, false, bh, (short)0, acc, false, false);
            }
            // C layout: VGPR j -> row (j + 8*kh), col = lane%16
            #pragma unroll
            for (int j = 0; j < 8; ++j) {
                sS[(j + 8 * kh) * SS_STRIDE + k0 + m] = acc[j] * SCALE;
            }
        }
    }
    __syncthreads();

    // ------------- Phase 2: causal softmax over each of 16 rows -------------
    {
        const int r = tid >> 4;          // row 0..15
        const int j = tid & 15;          // 16 threads per row
        const int q = q0 + r;            // last visible column (inclusive)

        float lmax = -3.402823466e38f;
        for (int c = j; c <= q; c += 16)
            lmax = fmaxf(lmax, sS[r * SS_STRIDE + c]);
        sRed[r * 16 + j] = lmax;
        __syncthreads();
        if (j == 0) {
            float mr = sRed[r * 16];
            #pragma unroll
            for (int t = 1; t < 16; ++t) mr = fmaxf(mr, sRed[r * 16 + t]);
            rowMax[r] = mr;
        }
        __syncthreads();

        const float mr = rowMax[r];
        float lsum = 0.f;
        for (int c = j; c < SEQ; c += 16) {
            if (c <= q) {
                float e = __expf(sS[r * SS_STRIDE + c] - mr);
                sS[r * SS_STRIDE + c] = e;
                lsum += e;
            } else {
                sS[r * SS_STRIDE + c] = 0.f;   // masked: exp(-1e9-m) == 0 in f32
            }
        }
        sRed[r * 16 + j] = lsum;
        __syncthreads();
        if (j == 0) {
            float s = 0.f;
            #pragma unroll
            for (int t = 0; t < 16; ++t) s += sRed[r * 16 + t];
            rowInv[r] = 1.f / s;
        }
        __syncthreads();
    }

    // ------- Phase 3: normalize in LDS + coalesced attention store ----------
    {
        float* attnBase = attn + ((size_t)b * SEQ + q0) * SEQ;
        for (int idx = tid; idx < 16 * SEQ; idx += 256) {
            const int r2 = idx >> 11;            // SEQ = 2048
            const int c2 = idx & (SEQ - 1);
            const float p = sS[r2 * SS_STRIDE + c2] * rowInv[r2];
            sS[r2 * SS_STRIDE + c2] = p;
            attnBase[(size_t)r2 * SEQ + c2] = p;
        }
    }
    __syncthreads();

    // ------- Phase 4: O = P V, exact f32 WMMA (wave owns 32 D-columns) ------
    {
        const int d0 = wave * 32;
        const float* Vb0 = V + (size_t)b * SEQ * DIM;

        v8f accA = {0.f, 0.f, 0.f, 0.f, 0.f, 0.f, 0.f, 0.f};
        v8f accB = {0.f, 0.f, 0.f, 0.f, 0.f, 0.f, 0.f, 0.f};

        for (int kt = 0; kt <= qTile; ++kt) {
            const int kbase = kt * 16;
            const float* Vb = Vb0 + (size_t)kbase * DIM;
            #pragma unroll
            for (int kk = 0; kk < 4; ++kk) {
                const int kr = kk * 4 + 2 * kh;   // this lane's two K rows
                v2f a = *(const v2f*)&sS[m * SS_STRIDE + kbase + kk * 4 + 2 * kh];
                v2f b0 = { Vb[(size_t)kr * DIM + d0 + m],
                           Vb[(size_t)(kr + 1) * DIM + d0 + m] };
                v2f b1 = { Vb[(size_t)kr * DIM + d0 + 16 + m],
                           Vb[(size_t)(kr + 1) * DIM + d0 + 16 + m] };
                accA = __builtin_amdgcn_wmma_f32_16x16x4_f32(
                    false, a, false, b0, (short)0, accA, false, false);
                accB = __builtin_amdgcn_wmma_f32_16x16x4_f32(
                    false, a, false, b1, (short)0, accB, false, false);
            }
        }

        float* ob = out + ((size_t)b * SEQ + q0) * DIM;
        #pragma unroll
        for (int j = 0; j < 8; ++j) {
            ob[(size_t)(j + 8 * kh) * DIM + d0 + m]      = accA[j];
            ob[(size_t)(j + 8 * kh) * DIM + d0 + 16 + m] = accB[j];
        }
    }
}

extern "C" void kernel_launch(void* const* d_in, const int* in_sizes, int n_in,
                              void* d_out, int out_size, void* d_ws, size_t ws_size,
                              hipStream_t stream) {
    const float* Q = (const float*)d_in[0];
    const float* K = (const float*)d_in[1];
    const float* V = (const float*)d_in[2];
    // d_in[3] is the causal mask; it is reproduced analytically in-kernel.

    float* out  = (float*)d_out;
    float* attn = out + (size_t)BATCH * SEQ * DIM;

    const size_t n = (size_t)BATCH * SEQ * DIM;          // 4,194,304 elements
    unsigned short* ws  = (unsigned short*)d_ws;          // 4n shorts = 33.5 MB
    unsigned short* Qhi = ws;
    unsigned short* Qlo = ws + n;
    unsigned short* Khi = ws + 2 * n;
    unsigned short* Klo = ws + 3 * n;

    const int n2 = (int)(n / 2);
    dim3 gsplit((n2 + 255) / 256);
    split_hi_lo_bf16<<<gsplit, 256, 0, stream>>>(Q, (unsigned*)Qhi, (unsigned*)Qlo, n2);
    split_hi_lo_bf16<<<gsplit, 256, 0, stream>>>(K, (unsigned*)Khi, (unsigned*)Klo, n2);

    const size_t shmem = (size_t)(16 * SS_STRIDE + 256 + 32) * sizeof(float);
    dim3 grid(BATCH * (SEQ / 16));
    sdpa_causal_wmma<<<grid, 256, shmem, stream>>>(Qhi, Qlo, Khi, Klo, V, out, attn);
}
